// Capsule_14551349199082
// MI455X (gfx1250) — matrix operand order repaired and checked by reading, not compiled, not run on hardware
//
#include <hip/hip_runtime.h>
#include <hip/hip_bf16.h>

// ---------------------------------------------------------------------------
// Sizes (from the reference): B=256, L=200, EMB=128, HCN=1000, HD=64, POI=20000
// Padded: L->224 (7 K-chunks of 32), HCN->1008 (63 M-tiles of 16), h K->1024
// ---------------------------------------------------------------------------
#define BN    256
#define LNUM  200
#define EMBD  128
#define HDIM  64
#define HCNN  1000
#define POIN  20000
#define LP    224
#define HP    1008
#define KH    1024

typedef __attribute__((ext_vector_type(16))) __bf16 v16bf;
typedef __attribute__((ext_vector_type(8)))  float  v8f;

union FragU { v16bf v; unsigned int u[8]; };

#define WMMA_BF16(a, b, c) \
  __builtin_amdgcn_wmma_f32_16x16x32_bf16(false, (a), false, (b), (short)0, (c), false, false)

// Hardware f32->bf16 conversion (v_cvt_*_bf16_f32 on gfx1250, RNE)
__device__ __forceinline__ unsigned short f2bf_u16(float f) {
  union { __bf16 h; unsigned short s; } r;
  r.h = (__bf16)f;
  return r.s;
}
__device__ __forceinline__ unsigned int pack_bf2(float lo, float hi) {
  union { __bf16 h[2]; unsigned int u; } r;
  r.h[0] = (__bf16)lo;
  r.h[1] = (__bf16)hi;
  return r.u;
}
__device__ __forceinline__ v8f zero8() {
  v8f z;
#pragma unroll
  for (int i = 0; i < 8; i++) z[i] = 0.f;
  return z;
}

// ---------------------------------------------------------------------------
// k0: zero padded bf16 buffers (u_map, h) and build padded f32 routing logits
// ---------------------------------------------------------------------------
__global__ void k0_init(const float* __restrict__ rw,
                        unsigned int* __restrict__ umap_dw,
                        unsigned int* __restrict__ h_dw,
                        float* __restrict__ brt) {
  int idx = blockIdx.x * blockDim.x + threadIdx.x;
  int stride = gridDim.x * blockDim.x;
  for (int i = idx; i < BN * LP * HDIM / 2; i += stride) umap_dw[i] = 0u;
  for (int i = idx; i < BN * KH / 2; i += stride) h_dw[i] = 0u;
  for (int i = idx; i < HP * LP; i += stride) {
    int h = i / LP, l = i % LP;
    brt[i] = (h < HCNN && l < LNUM) ? rw[h * LNUM + l] : 0.f;
  }
}

// ---------------------------------------------------------------------------
// k1: u_map = u @ l_map_h   ([B*L,128]x[128,64]) -> padded bf16 [B,224,64]
// l_map_h staged transposed ([n][k] bf16 pairs) in LDS for B-fragment loads.
// One wave = 16 rows x 64 cols; 400 blocks x 8 waves = 3200 row tiles exact.
// A fragments: 4x global_load_b128 (float4) per 32-K chunk.
// ---------------------------------------------------------------------------
__global__ void k1_umap(const float* __restrict__ u,
                        const float* __restrict__ lmap,
                        unsigned short* __restrict__ umap) {
  __shared__ unsigned int lds[64 * 64];  // [n=64][k/2=64] bf16 pairs, 16KB
  const int tid = threadIdx.x;
  for (int d = tid; d < 64 * 64; d += 256) {
    int n = d >> 6, jk = d & 63, k = jk << 1;
    lds[d] = pack_bf2(lmap[k * HDIM + n], lmap[(k + 1) * HDIM + n]);
  }
  __syncthreads();
  const int wave = tid >> 5, lane = tid & 31;
  const int half = lane >> 4, ln = lane & 15;
  const int tile = blockIdx.x * 8 + wave;  // 0..3199
  v8f acc[4];
#pragma unroll
  for (int t = 0; t < 4; t++) acc[t] = zero8();
  const int gA = tile * 16 + ln;  // A-row (M = lane&15)
  const float4* uA = reinterpret_cast<const float4*>(u + (size_t)gA * EMBD);
  for (int kk = 0; kk < EMBD; kk += 32) {
    FragU A;
    // group 0: K = kk + 8*half + 0..7 ; group 1: +16 (contiguous float8 each)
    int b0 = (kk + 8 * half) >> 2;
    float4 a0 = uA[b0], a1 = uA[b0 + 1], a2 = uA[b0 + 4], a3 = uA[b0 + 5];
    A.u[0] = pack_bf2(a0.x, a0.y); A.u[1] = pack_bf2(a0.z, a0.w);
    A.u[2] = pack_bf2(a1.x, a1.y); A.u[3] = pack_bf2(a1.z, a1.w);
    A.u[4] = pack_bf2(a2.x, a2.y); A.u[5] = pack_bf2(a2.z, a2.w);
    A.u[6] = pack_bf2(a3.x, a3.y); A.u[7] = pack_bf2(a3.z, a3.w);
#pragma unroll
    for (int t = 0; t < 4; t++) {
      FragU Bf;
      int n = t * 16 + ln;
#pragma unroll
      for (int j = 0; j < 8; j++) Bf.u[j] = lds[n * 64 + (kk >> 1) + 4 * half + (j & 3) + 8 * (j >> 2)];
      acc[t] = WMMA_BF16(A.v, Bf.v, acc[t]);
    }
  }
#pragma unroll
  for (int r = 0; r < 8; r++) {
    int grow = tile * 16 + r + 8 * half;  // D: M = r + 8*half
    int b = grow / LNUM, l = grow % LNUM;
    unsigned short* dst = umap + (b * LP + l) * HDIM + ln;
#pragma unroll
    for (int t = 0; t < 4; t++) dst[t * 16] = f2bf_u16(acc[t][r]);
  }
}

// ---------------------------------------------------------------------------
// k2: c = softmax(b, axis=-1) over L=200; write padded bf16 c [1008,224]
// ---------------------------------------------------------------------------
__global__ void k2_softmax(const float* __restrict__ brt,
                           unsigned short* __restrict__ cbf) {
  const int h = blockIdx.x;                  // 0..1007
  const int tid = threadIdx.x;               // 0..255
  const int lane = tid & 31, wave = tid >> 5;
  __shared__ float redmax[8], redsum[8];
  const bool act = (h < HCNN) && (tid < LNUM);
  float xv = act ? brt[h * LP + tid] : -3.0e38f;
  float m = xv;
  for (int s = 16; s >= 1; s >>= 1) m = fmaxf(m, __shfl_xor(m, s, 32));
  if (lane == 0) redmax[wave] = m;
  __syncthreads();
  float mx = redmax[0];
  for (int i = 1; i < 8; i++) mx = fmaxf(mx, redmax[i]);
  float e = act ? __expf(xv - mx) : 0.f;
  float s = e;
  for (int t = 16; t >= 1; t >>= 1) s += __shfl_xor(s, t, 32);
  if (lane == 0) redsum[wave] = s;
  __syncthreads();
  float tot = 0.f;
  for (int i = 0; i < 8; i++) tot += redsum[i];
  float inv = 1.f / tot;  // h>=1000 -> inf, masked by 'act' below
  if (tid < LP) cbf[h * LP + tid] = f2bf_u16(act ? e * inv : 0.f);
}

// ---------------------------------------------------------------------------
// k3: per batch b: s = c @ u_map[b]  ([1008,224]x[224,64]), squash -> v (bf16)
// last iteration: h[b,:] = v @ w_dim^T + b_dim directly (bf16, zero-padded)
// u_map[b] staged transposed in LDS (28KB). Grid (256, 8), 8 waves/block.
// ---------------------------------------------------------------------------
__global__ void k3_route(const unsigned short* __restrict__ umap,
                         const unsigned int* __restrict__ cbf_dw,
                         unsigned short* __restrict__ vbf,
                         unsigned short* __restrict__ hbf,
                         const float* __restrict__ wdim,
                         const float* __restrict__ bdim,
                         int compute_h) {
  __shared__ unsigned short lds_us[HDIM * LP];  // [n=64][k=224], 28KB
  const int tid = threadIdx.x;
  const int bId = blockIdx.x;
  const unsigned int* src = (const unsigned int*)(umap + bId * LP * HDIM);
  for (int sIdx = tid; sIdx < LP * (HDIM / 2); sIdx += 256) {
    int k = sIdx >> 5, t2 = sIdx & 31;  // source row-major [k][n] pairs in n
    unsigned int w = src[sIdx];
    lds_us[(2 * t2) * LP + k] = (unsigned short)(w & 0xFFFFu);
    lds_us[(2 * t2 + 1) * LP + k] = (unsigned short)(w >> 16);
  }
  __syncthreads();
  const unsigned int* lds_dw = (const unsigned int*)lds_us;
  const int wave = tid >> 5, lane = tid & 31, half = lane >> 4, ln = lane & 15;
  const int mtile = blockIdx.y * 8 + wave;
  if (mtile >= 63) return;
  v8f acc[4];
#pragma unroll
  for (int t = 0; t < 4; t++) acc[t] = zero8();
  const int mA = mtile * 16 + ln;
  for (int kk = 0; kk < 7; kk++) {  // 7 chunks of K=32 over Lp=224
    FragU A;
    {
      int base = mA * (LP / 2) + kk * 16 + 4 * half;
      const uint4* ca = reinterpret_cast<const uint4*>(cbf_dw + base);
      uint4 q0 = ca[0];                      // p = 0..3
      uint4 q1 = *reinterpret_cast<const uint4*>(cbf_dw + base + 8);  // p = 4..7
      A.u[0] = q0.x; A.u[1] = q0.y; A.u[2] = q0.z; A.u[3] = q0.w;
      A.u[4] = q1.x; A.u[5] = q1.y; A.u[6] = q1.z; A.u[7] = q1.w;
    }
#pragma unroll
    for (int t = 0; t < 4; t++) {
      FragU Bf;
      int n = t * 16 + ln;
#pragma unroll
      for (int j = 0; j < 8; j++)
        Bf.u[j] = lds_dw[n * (LP / 2) + kk * 16 + 4 * half + (j & 3) + 8 * (j >> 2)];
      acc[t] = WMMA_BF16(A.v, Bf.v, acc[t]);
    }
  }
  float wd[4];
#pragma unroll
  for (int t = 0; t < 4; t++) wd[t] = wdim[t * 16 + ln];
  float b0 = bdim[0];
#pragma unroll
  for (int r = 0; r < 8; r++) {
    int m = mtile * 16 + r + 8 * half;  // D: M = r + 8*half
    float sv[4];
    float sq = 0.f;
#pragma unroll
    for (int t = 0; t < 4; t++) { sv[t] = acc[t][r]; sq += sv[t] * sv[t]; }
    for (int s = 8; s >= 1; s >>= 1) sq += __shfl_xor(sq, s, 32);  // 16-lane row sum
    float scale = (m < HCNN) ? (sq / (1.f + sq)) * rsqrtf(sq + 1e-8f) : 0.f;
    if (!compute_h) {
#pragma unroll
      for (int t = 0; t < 4; t++)
        vbf[(bId * HP + m) * HDIM + t * 16 + ln] = f2bf_u16(scale * sv[t]);
    } else {
      float hp = 0.f;
#pragma unroll
      for (int t = 0; t < 4; t++) hp += scale * sv[t] * wd[t];
      for (int s = 8; s >= 1; s >>= 1) hp += __shfl_xor(hp, s, 32);
      if (ln == 0) hbf[bId * KH + m] = f2bf_u16((m < HCNN) ? (hp + b0) : 0.f);
    }
  }
}

// ---------------------------------------------------------------------------
// k4: b += einsum('bhd,bld->hl', v, u_map): GEMM [1008 x 16384] x [16384 x 224]
// K = (batch,d) flattened; uint4 (b128) fragment loads; working set sits in L2.
// One wave = 16x32 output tile; 441 waves (63 mtiles x 7 npairs) in 56 blocks.
// ---------------------------------------------------------------------------
__global__ void k4_uv(const unsigned int* __restrict__ vbf_dw,
                      const unsigned int* __restrict__ umap_dw,
                      float* __restrict__ brt) {
  const int tid = threadIdx.x;
  const int wave = tid >> 5, lane = tid & 31, half = lane >> 4, ln = lane & 15;
  const int wid = blockIdx.x * 8 + wave;
  if (wid >= 63 * 7) return;
  const int mtile = wid / 7, npair = wid % 7;
  v8f acc[2];
  acc[0] = zero8(); acc[1] = zero8();
  const int mA = mtile * 16 + ln;
  const int n0 = npair * 32 + ln;
  for (int kc = 0; kc < 512; kc++) {  // 512 chunks of K=32 over B*HD=16384
    int bb = kc >> 1;
    int dhalf = (kc & 1) << 4;  // dword offset of d block (0 or 16)
    FragU A;
    {
      int base = (bb * HP + mA) * (HDIM / 2) + dhalf + 4 * half;
      uint4 q0 = *reinterpret_cast<const uint4*>(vbf_dw + base);      // p=0..3
      uint4 q1 = *reinterpret_cast<const uint4*>(vbf_dw + base + 8);  // p=4..7
      A.u[0] = q0.x; A.u[1] = q0.y; A.u[2] = q0.z; A.u[3] = q0.w;
      A.u[4] = q1.x; A.u[5] = q1.y; A.u[6] = q1.z; A.u[7] = q1.w;
    }
#pragma unroll
    for (int t = 0; t < 2; t++) {
      FragU Bf;
      int base = (bb * LP + n0 + 16 * t) * (HDIM / 2) + dhalf + 4 * half;
      uint4 q0 = *reinterpret_cast<const uint4*>(umap_dw + base);      // j=0..3
      uint4 q1 = *reinterpret_cast<const uint4*>(umap_dw + base + 8);  // j=4..7
      Bf.u[0] = q0.x; Bf.u[1] = q0.y; Bf.u[2] = q0.z; Bf.u[3] = q0.w;
      Bf.u[4] = q1.x; Bf.u[5] = q1.y; Bf.u[6] = q1.z; Bf.u[7] = q1.w;
      acc[t] = WMMA_BF16(A.v, Bf.v, acc[t]);
    }
  }
#pragma unroll
  for (int t = 0; t < 2; t++)
#pragma unroll
    for (int r = 0; r < 8; r++) {
      int m = mtile * 16 + r + 8 * half;
      int n = npair * 32 + t * 16 + ln;
      brt[m * LP + n] += acc[t][r];  // each (m,n) owned by exactly one lane
    }
}

// ---------------------------------------------------------------------------
// k5: out = h @ w_part^T + b_part  ([256,1024]x[1024,20000])
// Block: M=128 (LDS-staged bf16 h chunk, ds_load_b128 A-fragments), N=128
// (8 waves x 16 w_part columns; f32 -> bf16 via hw cvt; K<1000 masked).
// Grid (157, 2): each w_part element read twice total.
// ---------------------------------------------------------------------------
__global__ void k5_out(const unsigned int* __restrict__ h_dw,
                       const float* __restrict__ wpart,
                       const float* __restrict__ bpart,
                       float* __restrict__ out) {
  __shared__ unsigned int hs[128 * 16];  // [mlocal][k/2] per 32-K chunk, 8KB
  const int tid = threadIdx.x, wave = tid >> 5, lane = tid & 31;
  const int half = lane >> 4, ln = lane & 15;
  const int pbase = blockIdx.x * 128 + wave * 16;
  const int mbase = blockIdx.y * 128;
  const int p = pbase + ln;
  const bool pv = p < POIN;
  const float bp = pv ? bpart[p] : 0.f;
  v8f acc[8];
#pragma unroll
  for (int t = 0; t < 8; t++) acc[t] = zero8();
  for (int kc = 0; kc < 32; kc++) {
    int kk = kc << 5;
    __syncthreads();  // previous-iteration A reads done before restage
    for (int d = tid; d < 128 * 16; d += 256) {
      int ml = d >> 4, kd = d & 15;
      int k = kk + 2 * kd;
      hs[d] = (k < HP) ? h_dw[(mbase + ml) * (KH / 2) + (k >> 1)] : 0u;
    }
    __syncthreads();
    if (pv && kk + 32 < HCNN)
      __builtin_prefetch(&wpart[(size_t)p * HCNN + kk + 32], 0, 0);
    FragU Bf;
    const int kb = kk + 16 * half;  // B K-base: floats kb .. kb+15 contiguous
    if (pv && kb + 15 < HCNN) {
      const float4* wp4 = reinterpret_cast<const float4*>(wpart + (size_t)p * HCNN + kb);
      float4 w0 = wp4[0], w1 = wp4[1], w2 = wp4[2], w3 = wp4[3];
      Bf.u[0] = pack_bf2(w0.x, w0.y); Bf.u[1] = pack_bf2(w0.z, w0.w);
      Bf.u[2] = pack_bf2(w1.x, w1.y); Bf.u[3] = pack_bf2(w1.z, w1.w);
      Bf.u[4] = pack_bf2(w2.x, w2.y); Bf.u[5] = pack_bf2(w2.z, w2.w);
      Bf.u[6] = pack_bf2(w3.x, w3.y); Bf.u[7] = pack_bf2(w3.z, w3.w);
    } else {
#pragma unroll
      for (int j = 0; j < 8; j++) {
        int k = kb + 2 * j;
        float f0 = (pv && k < HCNN) ? wpart[(size_t)p * HCNN + k] : 0.f;
        float f1 = (pv && (k + 1) < HCNN) ? wpart[(size_t)p * HCNN + k + 1] : 0.f;
        Bf.u[j] = pack_bf2(f0, f1);
      }
    }
#pragma unroll
    for (int mt = 0; mt < 8; mt++) {
      FragU A;
      int ml = mt * 16 + ln;
#pragma unroll
      for (int pp = 0; pp < 8; pp++)
        A.u[pp] = hs[ml * 16 + 8 * half + pp];
      acc[mt] = WMMA_BF16(A.v, Bf.v, acc[mt]);
    }
  }
  if (pv) {
#pragma unroll
    for (int mt = 0; mt < 8; mt++)
#pragma unroll
      for (int r = 0; r < 8; r++) {
        int row = mbase + mt * 16 + r + 8 * half;
        out[(size_t)row * POIN + p] = acc[mt][r] + bp;
      }
  }
}

// ---------------------------------------------------------------------------
// Workspace layout (bytes, 256-aligned): total ~42.25 MB
// ---------------------------------------------------------------------------
#define WS_UMAP 0u               // B*LP*HD bf16     = 7,340,032
#define WS_CBF  7340032u         // HP*LP bf16       =   451,584
#define WS_BRT  7791616u         // HP*LP f32        =   903,168
#define WS_VBF  8694784u         // B*HP*HD bf16     = 33,030,144
#define WS_HBF  41724928u        // B*KH bf16        =   524,288

extern "C" void kernel_launch(void* const* d_in, const int* in_sizes, int n_in,
                              void* d_out, int out_size, void* d_ws, size_t ws_size,
                              hipStream_t stream) {
  (void)in_sizes; (void)n_in; (void)out_size; (void)ws_size;
  const float* u     = (const float*)d_in[0];
  const float* lmap  = (const float*)d_in[1];
  const float* rw    = (const float*)d_in[2];
  const float* wdim  = (const float*)d_in[3];
  const float* bdim  = (const float*)d_in[4];
  const float* wpart = (const float*)d_in[5];
  const float* bpart = (const float*)d_in[6];
  // d_in[7] = rn (device int, == 3 in setup); routing count fixed below.
  float* out = (float*)d_out;

  char* ws = (char*)d_ws;
  unsigned short* umap    = (unsigned short*)(ws + WS_UMAP);
  unsigned int*   umap_dw = (unsigned int*)(ws + WS_UMAP);
  unsigned short* cbf     = (unsigned short*)(ws + WS_CBF);
  unsigned int*   cbf_dw  = (unsigned int*)(ws + WS_CBF);
  float*          brt     = (float*)(ws + WS_BRT);
  unsigned short* vbf     = (unsigned short*)(ws + WS_VBF);
  unsigned int*   vbf_dw  = (unsigned int*)(ws + WS_VBF);
  unsigned short* hbf     = (unsigned short*)(ws + WS_HBF);
  unsigned int*   h_dw    = (unsigned int*)(ws + WS_HBF);

  const int RN = 3;
  k0_init<<<512, 256, 0, stream>>>(rw, umap_dw, h_dw, brt);
  k1_umap<<<400, 256, 0, stream>>>(u, lmap, umap);
  for (int it = 0; it < RN; ++it) {
    k2_softmax<<<HP, 256, 0, stream>>>(brt, cbf);
    k3_route<<<dim3(BN, 8), 256, 0, stream>>>(umap, cbf_dw, vbf, hbf, wdim, bdim,
                                              (it == RN - 1) ? 1 : 0);
    if (it < RN - 1) k4_uv<<<56, 256, 0, stream>>>(vbf_dw, umap_dw, brt);
  }
  k5_out<<<dim3(157, 2), 256, 0, stream>>>(h_dw, wpart, bpart, out);
}
